// GG_23175643530135
// MI455X (gfx1250) — matrix operand order repaired and checked by reading, not compiled, not run on hardware
//
#include <hip/hip_runtime.h>
#include <hip/hip_bf16.h>

typedef __attribute__((ext_vector_type(16))) __bf16 v16bf;
typedef __attribute__((ext_vector_type(8)))  float  v8f;

#define N_NODES 16384
#define FIN     256
#define FOUT    64

// ---------------------------------------------------------------------------
// Kernel 1: support = X @ W  (fp32 accumulate), stored TRANSPOSED in bf16:
//   supportT[n][k],  n in [0,64), k in [0,16384)
// so that the WMMA B-operand (per-lane contiguous K) is a single 32B load.
// Each block handles 4 rows of X (k) x all 64 output columns (n).
// ---------------------------------------------------------------------------
__global__ __launch_bounds__(256) void gcn_support(
    const float* __restrict__ x, const float* __restrict__ w,
    __bf16* __restrict__ supportT) {
  __shared__ float xs[4 * FIN];
  const int kbase = blockIdx.x * 4;
  for (int i = threadIdx.x; i < 4 * FIN; i += 256)
    xs[i] = x[(size_t)kbase * FIN + i];
  __syncthreads();

  const int n  = threadIdx.x & 63;   // output column
  const int kl = threadIdx.x >> 6;   // 0..3 local row
  const float* xr = &xs[kl * FIN];
  float acc = 0.f;
#pragma unroll 8
  for (int f = 0; f < FIN; ++f)
    acc = fmaf(xr[f], w[f * FOUT + n], acc);
  supportT[(size_t)n * N_NODES + (kbase + kl)] = (__bf16)acc;
}

// ---------------------------------------------------------------------------
// Kernel 2: out = (adj @ support) / degree + bias  via bf16 WMMA.
// One wave (32 lanes) owns a 16-row x 64-col output tile.
// A (16x32 bf16), same contiguous K as B:
//   lanes 0-15  : row = lane,    K = kb + {0..15}
//   lanes 16-31 : row = lane-16, K = kb + {16..31}
// B (32x16 bf16): lane n = lane&15, K = kb + (lane>>4)*16 + {0..15}  -> one
//   contiguous 32B read from supportT.
// Row degree = row-sum of binary adj, accumulated while streaming A.
// ---------------------------------------------------------------------------
__global__ __launch_bounds__(64) void gcn_aggregate(
    const float* __restrict__ adj, const __bf16* __restrict__ supportT,
    const float* __restrict__ bias, float* __restrict__ out) {
  const int lane = threadIdx.x & 31;
  const int wave = threadIdx.x >> 5;
  const int tile = blockIdx.x * 2 + wave;       // 1024 tiles of 16 rows
  const int row_base = tile * 16;
  const int lrow = lane & 15;
  const int lhi  = lane >> 4;                   // 0 or 1

  const float*  arow  = adj + (size_t)(row_base + lrow) * N_NODES + lhi * 16;
  const __bf16* bbase = supportT + (size_t)lrow * N_NODES + lhi * 16;

  v8f acc[4] = {};
  float deg = 0.f;

#pragma unroll 2
  for (int kb = 0; kb < N_NODES; kb += 32) {
    const float* ap = arow + kb;
    const float4 a0 = *(const float4*)(ap + 0);
    const float4 a1 = *(const float4*)(ap + 4);
    const float4 a2 = *(const float4*)(ap + 8);
    const float4 a3 = *(const float4*)(ap + 12);
    __builtin_prefetch(ap + 512, 0, 1);         // ~16 iterations ahead

    // degree (adj is exactly 0/1 -> sum == nonzero count)
    deg += (a0.x + a0.y) + (a0.z + a0.w) + (a1.x + a1.y) + (a1.z + a1.w) +
           (a2.x + a2.y) + (a2.z + a2.w) + (a3.x + a3.y) + (a3.z + a3.w);

    v16bf A;
    A[0]  = (__bf16)a0.x; A[1]  = (__bf16)a0.y;
    A[2]  = (__bf16)a0.z; A[3]  = (__bf16)a0.w;
    A[4]  = (__bf16)a1.x; A[5]  = (__bf16)a1.y;
    A[6]  = (__bf16)a1.z; A[7]  = (__bf16)a1.w;
    A[8]  = (__bf16)a2.x; A[9]  = (__bf16)a2.y;
    A[10] = (__bf16)a2.z; A[11] = (__bf16)a2.w;
    A[12] = (__bf16)a3.x; A[13] = (__bf16)a3.y;
    A[14] = (__bf16)a3.z; A[15] = (__bf16)a3.w;

    const __bf16* bp = bbase + kb;
#pragma unroll
    for (int c = 0; c < 4; ++c) {
      const v16bf B = *(const v16bf*)(bp + (size_t)c * 16 * N_NODES);
      acc[c] = __builtin_amdgcn_wmma_f32_16x16x32_bf16(
          false, A, false, B, (short)0, acc[c], false, false);
    }
  }

  // combine the two half-lane partial degrees: lane L and L^16 cover row L&15
  const float degsum = deg + __shfl_xor(deg, 16, 32);

  float bv[4];
#pragma unroll
  for (int c = 0; c < 4; ++c) bv[c] = bias[c * 16 + lrow];

  const int hi8 = lhi * 8;
#pragma unroll
  for (int v = 0; v < 8; ++v) {
    const float d   = __shfl(degsum, v + hi8, 32);  // degree(row v+hi8)
    const float inv = 1.0f / d;
    const int   row = row_base + hi8 + v;           // C layout: M = v + 8*lhi
    float* orow = out + (size_t)row * FOUT + lrow;
#pragma unroll
    for (int c = 0; c < 4; ++c)
      orow[c * 16] = acc[c][v] * inv + bv[c];
  }
}

// ---------------------------------------------------------------------------
extern "C" void kernel_launch(void* const* d_in, const int* in_sizes, int n_in,
                              void* d_out, int out_size, void* d_ws, size_t ws_size,
                              hipStream_t stream) {
  const float* x    = (const float*)d_in[0];   // [16384, 256]
  const float* adj  = (const float*)d_in[1];   // [16384, 16384]
  const float* w    = (const float*)d_in[2];   // [256, 64]
  const float* bias = (const float*)d_in[3];   // [64]
  float* out = (float*)d_out;                  // [16384, 64]

  __bf16* supportT = (__bf16*)d_ws;            // 64 * 16384 * 2B = 2 MB

  gcn_support<<<N_NODES / 4, 256, 0, stream>>>(x, w, supportT);
  gcn_aggregate<<<N_NODES / 32, 64, 0, stream>>>(adj, supportT, bias, out);
}
